// QuantizedLinear_89747636617907
// MI455X (gfx1250) — compile-verified, matrix operand order checked by reading, not run on hardware
//
#include <hip/hip_runtime.h>

typedef __bf16 bf16_t;
typedef __attribute__((ext_vector_type(16))) __bf16 v16bf;
typedef __attribute__((ext_vector_type(8)))  float  v8f;

#define M_TILE 128
#define N_TILE 128
#define K_TILE 32
#define LDS_STRIDE 40   // bf16 elems/row: 32 data + 8 pad = 80 B (bank-conflict-free, 16B aligned)

#define K_DIM 4096
#define NIG   512                    // in-groups per output row (4096 / 8)
#define NITER (K_DIM / K_TILE)       // 128

// Per-thread register staging for one K tile (A: 16 floats, B: 2 codebook entries)
struct StageRegs {
    float4 a[4];
    float4 c[2][2];
};

__device__ __forceinline__ void stage_load(const float* __restrict__ aptr,
                                           const int*   __restrict__ cptr,   // pre-offset by gbase
                                           const float* __restrict__ cbooks,
                                           int kk, StageRegs& r) {
    const float4* ap = (const float4*)(aptr + kk);
    r.a[0] = ap[0]; r.a[1] = ap[1]; r.a[2] = ap[2]; r.a[3] = ap[3];
    const int kg = kk >> 3;
    #pragma unroll
    for (int g = 0; g < 2; ++g) {
        const int code = cptr[kg + g];
        const float4* cb = (const float4*)(cbooks + (size_t)code * 8);
        r.c[g][0] = cb[0];
        r.c[g][1] = cb[1];
    }
}

__device__ __forceinline__ void stage_store(bf16_t* __restrict__ bufA,
                                            bf16_t* __restrict__ bufB,
                                            int arow, int aoff, int brow, int gbase,
                                            float scale, const StageRegs& r) {
    union { bf16_t h[16]; uint4 q[2]; } au;
    au.h[0]  = (bf16_t)r.a[0].x; au.h[1]  = (bf16_t)r.a[0].y;
    au.h[2]  = (bf16_t)r.a[0].z; au.h[3]  = (bf16_t)r.a[0].w;
    au.h[4]  = (bf16_t)r.a[1].x; au.h[5]  = (bf16_t)r.a[1].y;
    au.h[6]  = (bf16_t)r.a[1].z; au.h[7]  = (bf16_t)r.a[1].w;
    au.h[8]  = (bf16_t)r.a[2].x; au.h[9]  = (bf16_t)r.a[2].y;
    au.h[10] = (bf16_t)r.a[2].z; au.h[11] = (bf16_t)r.a[2].w;
    au.h[12] = (bf16_t)r.a[3].x; au.h[13] = (bf16_t)r.a[3].y;
    au.h[14] = (bf16_t)r.a[3].z; au.h[15] = (bf16_t)r.a[3].w;
    *(uint4*)&bufA[arow * LDS_STRIDE + aoff]     = au.q[0];
    *(uint4*)&bufA[arow * LDS_STRIDE + aoff + 8] = au.q[1];

    #pragma unroll
    for (int g = 0; g < 2; ++g) {
        union { bf16_t h[8]; uint4 q; } bu;
        bu.h[0] = (bf16_t)(r.c[g][0].x * scale); bu.h[1] = (bf16_t)(r.c[g][0].y * scale);
        bu.h[2] = (bf16_t)(r.c[g][0].z * scale); bu.h[3] = (bf16_t)(r.c[g][0].w * scale);
        bu.h[4] = (bf16_t)(r.c[g][1].x * scale); bu.h[5] = (bf16_t)(r.c[g][1].y * scale);
        bu.h[6] = (bf16_t)(r.c[g][1].z * scale); bu.h[7] = (bf16_t)(r.c[g][1].w * scale);
        *(uint4*)&bufB[brow * LDS_STRIDE + (gbase + g) * 8] = bu.q;
    }
}

// Load A/B fragments per ISA 16-bit operand layout and run the 4x2 WMMA grid.
__device__ __forceinline__ void compute_tile(const bf16_t* __restrict__ bufA,
                                             const bf16_t* __restrict__ bufB,
                                             int wm, int wn, int half, int l15,
                                             v8f acc[4][2]) {
    union frag_u { uint4 q[2]; v16bf v; };
    v16bf afr[4];
    #pragma unroll
    for (int tm = 0; tm < 4; ++tm) {
        const int row = wm * 64 + tm * 16 + l15;
        frag_u u;
        u.q[0] = *(const uint4*)&bufA[row * LDS_STRIDE + half * 8];
        u.q[1] = *(const uint4*)&bufA[row * LDS_STRIDE + half * 8 + 16];
        afr[tm] = u.v;
    }
    v16bf bfr[2];
    #pragma unroll
    for (int tn = 0; tn < 2; ++tn) {
        const int row = wn * 32 + tn * 16 + l15;
        frag_u u;
        u.q[0] = *(const uint4*)&bufB[row * LDS_STRIDE + half * 8];
        u.q[1] = *(const uint4*)&bufB[row * LDS_STRIDE + half * 8 + 16];
        bfr[tn] = u.v;
    }
    #pragma unroll
    for (int tm = 0; tm < 4; ++tm)
        #pragma unroll
        for (int tn = 0; tn < 2; ++tn)
            acc[tm][tn] = __builtin_amdgcn_wmma_f32_16x16x32_bf16(
                false, afr[tm], false, bfr[tn],
                (short)0, acc[tm][tn], false, false);
}

__global__ __launch_bounds__(256) void aqlm_wmma_gemm(
    const float* __restrict__ A,        // [M, 4096] f32
    const int*   __restrict__ codes,    // [4096, 512] int32
    const float* __restrict__ cbooks,   // [65536, 8] f32
    const float* __restrict__ scales,   // [4096] f32
    const float* __restrict__ bias,     // [4096] f32
    float*       __restrict__ out,      // [M, 4096] f32
    int M, int N)
{
    // Double-buffered tiles: 2 * (10 KB + 10 KB) = 40 KB of the WGP's 320 KB LDS
    __shared__ bf16_t lA[2][M_TILE * LDS_STRIDE];
    __shared__ bf16_t lB[2][N_TILE * LDS_STRIDE];

    const int tid  = threadIdx.x;
    const int lane = tid & 31;
    const int wave = tid >> 5;     // 0..7
    const int wm   = wave >> 2;    // 0..1 -> 64-row band
    const int wn   = wave & 3;     // 0..3 -> 32-col band
    const int half = lane >> 4;    // 0/1 per WMMA 16-bit operand layout
    const int l15  = lane & 15;

    const int m0 = blockIdx.y * M_TILE;
    const int n0 = blockIdx.x * N_TILE;

    // A staging: 2 threads per row, 16 consecutive floats each
    const int arow = tid >> 1;
    const int aoff = (tid & 1) * 16;
    // B staging: 2 threads per output row, 2 in-groups each
    const int brow  = tid >> 1;
    const int gbase = (tid & 1) * 2;

    const float scale = scales[n0 + brow];
    const float* aptr = A + (size_t)(m0 + arow) * K_DIM + aoff;
    const int*   cptr = codes + (size_t)(n0 + brow) * NIG + gbase;

    v8f acc[4][2];
    #pragma unroll
    for (int i = 0; i < 4; ++i)
        #pragma unroll
        for (int j = 0; j < 2; ++j)
            acc[i][j] = (v8f){0.f,0.f,0.f,0.f,0.f,0.f,0.f,0.f};

    // Prologue: fill buffer 0
    StageRegs r;
    stage_load(aptr, cptr, cbooks, 0, r);
    stage_store(lA[0], lB[0], arow, aoff, brow, gbase, scale, r);

    // Main loop: one barrier per iteration; stage next tile while WMMAs run
    for (int it = 0; it < NITER - 1; ++it) {
        __syncthreads();                                   // buf[it&1] ready
        const int cur = it & 1;
        stage_load(aptr, cptr, cbooks, (it + 1) * K_TILE, r);
        // writes of buf[cur^1] are issued before fragment loads (reads of buf[cur])
        stage_store(lA[cur ^ 1], lB[cur ^ 1], arow, aoff, brow, gbase, scale, r);
        compute_tile(lA[cur], lB[cur], wm, wn, half, l15, acc);
    }
    __syncthreads();
    compute_tile(lA[(NITER - 1) & 1], lB[(NITER - 1) & 1], wm, wn, half, l15, acc);

    // Epilogue: + bias, store f32 (one 64-bit base per tile, int offsets per row)
    #pragma unroll
    for (int tm = 0; tm < 4; ++tm) {
        #pragma unroll
        for (int tn = 0; tn < 2; ++tn) {
            const int n  = n0 + wn * 32 + tn * 16 + l15;
            const float bv = bias[n];
            float* po = out + (size_t)(m0 + wm * 64 + tm * 16 + half * 8) * N + n;
            #pragma unroll
            for (int i = 0; i < 8; ++i) {
                po[i * N] = acc[tm][tn][i] + bv;
            }
        }
    }
}

extern "C" void kernel_launch(void* const* d_in, const int* in_sizes, int n_in,
                              void* d_out, int out_size, void* d_ws, size_t ws_size,
                              hipStream_t stream) {
    const float* A      = (const float*)d_in[0];   // [4,2048,4096] f32
    const int*   codes  = (const int*)  d_in[1];   // [4096,512,1]
    const float* cbooks = (const float*)d_in[2];   // [1,65536,1,8]
    const float* scales = (const float*)d_in[3];   // [4096,1,1,1]
    const float* bias   = (const float*)d_in[4];   // [4096]
    float*       out    = (float*)d_out;

    const int K = K_DIM;
    const int N = in_sizes[4];            // 4096
    const int M = in_sizes[0] / K;        // 8192

    dim3 grid(N / N_TILE, M / M_TILE);    // (32, 64)
    dim3 block(256);
    aqlm_wmma_gemm<<<grid, block, 0, stream>>>(A, codes, cbooks, scales, bias, out, M, N);
}
